// TestSiluMul_44401371906471
// MI455X (gfx1250) — 8-bit matrix operand order repaired and checked by reading, compiled, not run on hardware
//
#include <hip/hip_runtime.h>
#include <hip/hip_bf16.h>


#define HIDDEN 4096
#define MDIM   4096
#define NDIM   4096
#define KDIM   4096
#define FP8_MAX 448.0f

typedef __attribute__((ext_vector_type(16))) int   v16i;
typedef __attribute__((ext_vector_type(8)))  float v8f;
typedef __attribute__((ext_vector_type(4)))  int   v4i;
typedef unsigned long long u64;

__device__ __forceinline__ float silu_f(float g) {
    return g / (1.0f + __expf(-g));
}
__device__ __forceinline__ float clamp_fp8(float v) {
    return fminf(fmaxf(v, -FP8_MAX), FP8_MAX);
}

__global__ __launch_bounds__(256)
void silu_mul_quant_kernel(const float* __restrict__ x,
                           const float* __restrict__ wscale,
                           unsigned char* __restrict__ Aq) {
    const int t    = blockIdx.x * blockDim.x + threadIdx.x;
    const int row  = t >> 10;
    const int col4 = (t & 1023) << 2;

    const float* g = x + (size_t)row * (2 * HIDDEN) + col4;
    const float* u = g + HIDDEN;
    float4 gv = *(const float4*)g;
    float4 uv = *(const float4*)u;

    const float s = wscale[0];

    float y0 = clamp_fp8((silu_f(gv.x) * uv.x) / s);
    float y1 = clamp_fp8((silu_f(gv.y) * uv.y) / s);
    float y2 = clamp_fp8((silu_f(gv.z) * uv.z) / s);
    float y3 = clamp_fp8((silu_f(gv.w) * uv.w) / s);

    int packed = __builtin_amdgcn_cvt_pk_fp8_f32(y0, y1, 0, false);
    packed     = __builtin_amdgcn_cvt_pk_fp8_f32(y2, y3, packed, true);

    *(int*)(Aq + (size_t)row * KDIM + col4) = packed;
}

__global__ __launch_bounds__(256)
void w_transpose_quant_kernel(const float* __restrict__ w,
                              unsigned char* __restrict__ Bt) {
    const int n  = blockIdx.x * 256 + threadIdx.x;
    const int k0 = blockIdx.y * 4;

    float w0 = w[(size_t)(k0 + 0) * NDIM + n];
    float w1 = w[(size_t)(k0 + 1) * NDIM + n];
    float w2 = w[(size_t)(k0 + 2) * NDIM + n];
    float w3 = w[(size_t)(k0 + 3) * NDIM + n];

    int packed = __builtin_amdgcn_cvt_pk_fp8_f32(w0, w1, 0, false);
    packed     = __builtin_amdgcn_cvt_pk_fp8_f32(w2, w3, packed, true);

    *(int*)(Bt + (size_t)n * KDIM + k0) = packed;
}

#define KTILE      128
#define NKSTEPS    (KDIM / KTILE)
#define LDS_STRIDE 144
#define TILE_BYTES (128 * LDS_STRIDE)

__device__ __forceinline__ void async_cp16(unsigned char* lds,
                                           const unsigned char* g) {
    unsigned lds_off = (unsigned)(size_t)lds;
    unsigned long long ga = (unsigned long long)(size_t)g;
    asm volatile("global_load_async_to_lds_b128 %0, %1, off"
                 :
                 : "v"(lds_off), "v"(ga), "v"(lds)
                 : "memory");
}

__device__ __forceinline__ v16i load_a_frag_lds(const unsigned char* base) {
    v16i a;
    u64* ap = (u64*)&a;
#pragma unroll
    for (int c = 0; c < 4; ++c) {
        ap[c]     = *(const u64*)(base + c * 16);
        ap[c + 4] = *(const u64*)(base + 64 + c * 16);
    }
    return a;
}

__device__ __forceinline__ v16i load_b_frag_lds(const unsigned char* base) {
    v16i b;
    u64* bp = (u64*)&b;
#pragma unroll
    for (int c = 0; c < 4; ++c) {
        bp[c]     = *(const u64*)(base + c * 16);
        bp[c + 4] = *(const u64*)(base + 64 + c * 16);
    }
    return b;
}

__global__ __launch_bounds__(256)
void fp8_gemm_kernel(const unsigned char* __restrict__ A,
                     const unsigned char* __restrict__ Bt,
                     const float* __restrict__ wscale,
                     float* __restrict__ out) {
    __shared__ __align__(16) unsigned char smem[4 * TILE_BYTES];

    const int tid  = threadIdx.x;
    const int lane = tid & 31;
    const int wave = tid >> 5;
    const int wm   = wave & 3;
    const int wn   = wave >> 2;

    const int m0g = blockIdx.x * 128;
    const int n0g = blockIdx.y * 128;

    const int l16  = lane & 15;
    const int half = lane >> 4;

    auto stage_tile = [&](int buf, int k0) {
#pragma unroll
        for (int c = 0; c < 4; ++c) {
            const int idx = tid + c * 256;
            const int row = idx >> 3;
            const int col = (idx & 7) << 4;
            async_cp16(smem + buf * TILE_BYTES + row * LDS_STRIDE + col,
                       A + (size_t)(m0g + row) * KDIM + k0 + col);
        }
#pragma unroll
        for (int c = 0; c < 4; ++c) {
            const int idx = tid + c * 256;
            const int row = idx >> 3;
            const int col = (idx & 7) << 4;
            async_cp16(smem + (2 + buf) * TILE_BYTES + row * LDS_STRIDE + col,
                       Bt + (size_t)(n0g + row) * KDIM + k0 + col);
        }
    };

    v8f acc[2][4] = {};

    stage_tile(0, 0);

    for (int ks = 0; ks < NKSTEPS; ++ks) {
        const int cur = ks & 1;

        if (ks + 1 < NKSTEPS) {
            stage_tile(cur ^ 1, (ks + 1) * KTILE);
            asm volatile("s_wait_asynccnt 0x8" ::: "memory");
        } else {
            asm volatile("s_wait_asynccnt 0x0" ::: "memory");
        }
        __syncthreads();

        const unsigned char* As = smem + cur * TILE_BYTES;
        const unsigned char* Bs = smem + (2 + cur) * TILE_BYTES;

        v16i a0 = load_a_frag_lds(As + (wm * 32 + 0  + l16) * LDS_STRIDE + half * 8);
        v16i a1 = load_a_frag_lds(As + (wm * 32 + 16 + l16) * LDS_STRIDE + half * 8);

#pragma unroll
        for (int tn = 0; tn < 4; ++tn) {
            v16i b = load_b_frag_lds(Bs + (wn * 64 + tn * 16 + l16) * LDS_STRIDE
                                        + half * 8);
            acc[0][tn] = __builtin_amdgcn_wmma_f32_16x16x128_fp8_fp8(
                a0, b, (short)0, acc[0][tn], false, false);
            acc[1][tn] = __builtin_amdgcn_wmma_f32_16x16x128_fp8_fp8(
                a1, b, (short)0, acc[1][tn], false, false);
        }

        __syncthreads();
    }

    const float s  = wscale[0];
    const float s2 = s * s;

    const int m0 = m0g + wm * 32;
    const int n0 = n0g + wn * 64;

#pragma unroll
    for (int tm = 0; tm < 2; ++tm) {
#pragma unroll
        for (int tn = 0; tn < 4; ++tn) {
            const int row_base = m0 + tm * 16 + half * 8;
            const int col      = n0 + tn * 16 + l16;
#pragma unroll
            for (int r = 0; r < 8; ++r) {
                out[(size_t)(row_base + r) * NDIM + col] = acc[tm][tn][r] * s2;
            }
        }
    }
}

extern "C" void kernel_launch(void* const* d_in, const int* in_sizes, int n_in,
                              void* d_out, int out_size, void* d_ws, size_t ws_size,
                              hipStream_t stream) {
    const float* x      = (const float*)d_in[0];
    const float* w      = (const float*)d_in[1];
    const float* wscale = (const float*)d_in[2];
    float* out          = (float*)d_out;

    unsigned char* Aq = (unsigned char*)d_ws;
    unsigned char* Bt = Aq + (size_t)MDIM * KDIM;

    {
        dim3 grid((MDIM * KDIM / 4) / 256);
        silu_mul_quant_kernel<<<grid, 256, 0, stream>>>(x, wscale, Aq);
    }

    {
        dim3 grid(NDIM / 256, KDIM / 4);
        w_transpose_quant_kernel<<<grid, 256, 0, stream>>>(w, Bt);
    }

    {
        dim3 grid(MDIM / 128, NDIM / 128);
        fp8_gemm_kernel<<<grid, 256, 0, stream>>>(Aq, Bt, wscale, out);
    }
}
